// GATGraphLearner_26517128086121
// MI455X (gfx1250) — matrix operand order repaired and checked by reading, not compiled, not run on hardware
//
#include <hip/hip_runtime.h>
#include <hip/hip_bf16.h>

#define NG 16
#define AN 1024
#define DF 256
#define NEG_SLOPE 0.2f

typedef __attribute__((ext_vector_type(16))) _Float16 v16h;
typedef __attribute__((ext_vector_type(8)))  float    v8f;

// ---------------------------------------------------------------------------
// Kernel 1: xw[g][a] = dot(x[g][a][:], W[:,0]) via v_wmma_f32_16x16x32_f16.
// One wave per 16-node tile; K looped in chunks of 32.
// A layout (16-bit A 16x32): lanes 0-15: elems0-7=K0-7, elems8-15=K16-23;
//                            lanes16-31: elems0-7=K8-15, elems8-15=K24-31.
// B = W broadcast over all 16 columns: same K order as A (elems0-7=K0-7/K8-15, elems8-15=K16-23/K24-31).
// D: VGPR j, lane 0 -> D[j][0] (nodes 0..7), lane 16 -> D[8+j][0] (nodes 8..15).
// ---------------------------------------------------------------------------
__global__ void xw_wmma_kernel(const float* __restrict__ x,
                               const float* __restrict__ W,
                               float* __restrict__ xw) {
  const int wave  = (blockIdx.x * blockDim.x + threadIdx.x) >> 5;
  const int lane  = threadIdx.x & 31;
  const int graph = wave >> 6;              // 64 waves per graph (1024/16)
  const int nodeB = (wave & 63) << 4;       // node base of this 16-row tile
  const bool hi   = (lane >= 16);
  const int  m    = hi ? (lane - 16) : lane;

  const float* __restrict__ xrow = x + ((size_t)graph * AN + nodeB + m) * DF;

  v8f c = {};
  for (int k0 = 0; k0 < DF; k0 += 32) {
    v16h a, b;
    const int ka0 = k0 + (hi ? 8 : 0);        // elems 0..7
    const int ka1 = k0 + 16 + (hi ? 8 : 0);   // elems 8..15
#pragma unroll
    for (int e = 0; e < 8; ++e) {
      a[e]     = (_Float16)xrow[ka0 + e];
      a[8 + e] = (_Float16)xrow[ka1 + e];
    }
    const int kb = k0 + (hi ? 16 : 0);
#pragma unroll
    for (int e = 0; e < 16; ++e) b[e] = (_Float16)W[k0 + (hi ? 8 : 0) + (e < 8 ? e : 8 + e)];
    c = __builtin_amdgcn_wmma_f32_16x16x32_f16(
        /*neg_a=*/false, a, /*neg_b=*/false, b,
        /*c_mod=*/(short)0, c, /*reuse_a=*/false, /*reuse_b=*/false);
  }

  if ((lane & 15) == 0) {                     // lane 0 and lane 16
    float* dst = xw + (size_t)graph * AN + nodeB + (hi ? 8 : 0);
#pragma unroll
    for (int j = 0; j < 8; ++j) dst[j] = c[j];
  }
}

// ---------------------------------------------------------------------------
// Kernel 2: per-dst-column online softmax over incoming edges + tanh(out).
// One thread per destination column; adj reads coalesced across the block.
// ---------------------------------------------------------------------------
__global__ void col_softmax_kernel(const float* __restrict__ adj,
                                   const float* __restrict__ xw_g,
                                   const float* __restrict__ att_src,
                                   const float* __restrict__ att_dst,
                                   const float* __restrict__ bias,
                                   float* __restrict__ t_out) {
  __shared__ float sxw[AN];                   // 4 KB of the 320 KB LDS
  const int graph = blockIdx.x / (AN / 256);
  const int dtile = blockIdx.x % (AN / 256);
  const int d     = dtile * 256 + threadIdx.x;

  const float* __restrict__ xw = xw_g + (size_t)graph * AN;
  for (int s = threadIdx.x; s < AN; s += 256) sxw[s] = xw[s];
  __syncthreads();

  const float as_c = att_src[0];
  const float a_d  = sxw[d] * att_dst[0];
  const float* __restrict__ col = adj + (size_t)graph * AN * AN + d;

  float m = -INFINITY, sum = 0.f, z = 0.f;
  for (int s = 0; s < AN; ++s) {
    const float av = col[(size_t)s * AN];
    if (av > 0.f) {
      float e = sxw[s] * as_c + a_d;
      e = (e >= 0.f) ? e : NEG_SLOPE * e;
      if (e > m) {                            // online rescale (exp(-inf)=0)
        const float r = expf(m - e);
        sum *= r; z *= r; m = e;
      }
      const float p = expf(e - m);
      sum += p;
      z   += p * sxw[s];
    }
  }
  const float out = (sum > 0.f) ? (z / sum + bias[0]) : bias[0];
  t_out[(size_t)graph * AN + d] = tanhf(out);
}

// ---------------------------------------------------------------------------
// Kernel 3: per-row nonzero count + zero-fill pred_adj (fused).
// One block (256 threads, float4 each) per adjacency row.
// ---------------------------------------------------------------------------
__global__ void count_zero_kernel(const float* __restrict__ adj,
                                  float* __restrict__ pred_adj,
                                  int* __restrict__ row_cnt) {
  const int row = blockIdx.x;                 // 0 .. NG*AN-1
  const int t   = threadIdx.x;
  const float4 a4 = ((const float4*)(adj + (size_t)row * AN))[t];
  ((float4*)(pred_adj + (size_t)row * AN))[t] = make_float4(0.f, 0.f, 0.f, 0.f);

  int cnt = (a4.x > 0.f) + (a4.y > 0.f) + (a4.z > 0.f) + (a4.w > 0.f);
  __shared__ int sc[256];
  sc[t] = cnt; __syncthreads();
  for (int off = 128; off > 0; off >>= 1) {
    if (t < off) sc[t] += sc[t + off];
    __syncthreads();
  }
  if (t == 0) row_cnt[row] = sc[0];
}

// ---------------------------------------------------------------------------
// Kernel 4: per-graph exclusive prefix sum of the 1024 row counts.
// ---------------------------------------------------------------------------
__global__ void scan_kernel(const int* __restrict__ row_cnt,
                            int* __restrict__ row_off) {
  const int graph = blockIdx.x;
  const int t     = threadIdx.x;
  const int base  = graph * AN;

  int v[4], s = 0;
#pragma unroll
  for (int j = 0; j < 4; ++j) { v[j] = row_cnt[base + t * 4 + j]; s += v[j]; }

  __shared__ int part[256];
  part[t] = s; __syncthreads();
  for (int off = 1; off < 256; off <<= 1) {
    const int val = (t >= off) ? part[t - off] : 0;
    __syncthreads();
    part[t] += val;
    __syncthreads();
  }
  int run = (t == 0) ? 0 : part[t - 1];
#pragma unroll
  for (int j = 0; j < 4; ++j) { row_off[base + t * 4 + j] = run; run += v[j]; }
}

// ---------------------------------------------------------------------------
// Kernel 5: scatter tanh(out)[edge_ordinal] into pred_adj[src][dst].
// Ordinal = row_off[row] + (intra-row rank of the nonzero), via block scan.
// ---------------------------------------------------------------------------
__global__ void scatter_kernel(const float* __restrict__ adj,
                               const int* __restrict__ row_off,
                               const float* __restrict__ t_out,
                               float* __restrict__ pred_adj) {
  const int row   = blockIdx.x;               // 0 .. NG*AN-1
  const int graph = row / AN;
  const int t     = threadIdx.x;

  const float4 a4 = ((const float4*)(adj + (size_t)row * AN))[t];
  const int f0 = a4.x > 0.f, f1 = a4.y > 0.f, f2 = a4.z > 0.f, f3 = a4.w > 0.f;

  __shared__ int part[256];
  part[t] = f0 + f1 + f2 + f3; __syncthreads();
  for (int off = 1; off < 256; off <<= 1) {
    const int val = (t >= off) ? part[t - off] : 0;
    __syncthreads();
    part[t] += val;
    __syncthreads();
  }
  int ord = row_off[row] + ((t == 0) ? 0 : part[t - 1]);

  const float* __restrict__ tg = t_out + (size_t)graph * AN;
  float* __restrict__ prow = pred_adj + (size_t)row * AN;
  const int c = t * 4;
  if (f0) { prow[c + 0] = tg[min(ord, AN - 1)]; ++ord; }
  if (f1) { prow[c + 1] = tg[min(ord, AN - 1)]; ++ord; }
  if (f2) { prow[c + 2] = tg[min(ord, AN - 1)]; ++ord; }
  if (f3) { prow[c + 3] = tg[min(ord, AN - 1)]; ++ord; }
}

// ---------------------------------------------------------------------------
// Kernel 6: embeddings = x (vectorized copy).
// ---------------------------------------------------------------------------
__global__ void copy_kernel(const float4* __restrict__ src,
                            float4* __restrict__ dst, int n4) {
  const int i = blockIdx.x * blockDim.x + threadIdx.x;
  if (i < n4) dst[i] = src[i];
}

// ---------------------------------------------------------------------------
extern "C" void kernel_launch(void* const* d_in, const int* in_sizes, int n_in,
                              void* d_out, int out_size, void* d_ws, size_t ws_size,
                              hipStream_t stream) {
  const float* x        = (const float*)d_in[0];   // [NG, AN, DF]
  const float* adj      = (const float*)d_in[1];   // [NG, AN, AN]
  const float* W        = (const float*)d_in[2];   // [DF, 1]
  const float* att_src  = (const float*)d_in[3];   // [1]
  const float* att_dst  = (const float*)d_in[4];   // [1]
  const float* bias     = (const float*)d_in[5];   // [1]

  float* pred_adj = (float*)d_out;                       // NG*AN*AN floats
  float* emb      = (float*)d_out + (size_t)NG * AN * AN; // NG*AN*DF floats

  // workspace layout
  char* ws = (char*)d_ws;
  float* xw      = (float*)(ws);                           // NG*AN f32
  float* t_out   = (float*)(ws + 1 * NG * AN * 4);         // NG*AN f32
  int*   row_cnt = (int*)  (ws + 2 * NG * AN * 4);         // NG*AN i32
  int*   row_off = (int*)  (ws + 3 * NG * AN * 4);         // NG*AN i32

  // 1) xw = x @ W  (WMMA): NG*AN/16 = 1024 waves, 8 waves/block -> 128 blocks
  xw_wmma_kernel<<<dim3((NG * AN / 16) / 8), dim3(256), 0, stream>>>(x, W, xw);

  // 2) per-column online softmax + tanh: NG * (AN/256) = 64 blocks
  col_softmax_kernel<<<dim3(NG * (AN / 256)), dim3(256), 0, stream>>>(
      adj, xw, att_src, att_dst, bias, t_out);

  // 3) row counts + zero pred_adj: one block per row
  count_zero_kernel<<<dim3(NG * AN), dim3(256), 0, stream>>>(adj, pred_adj, row_cnt);

  // 4) exclusive scan per graph
  scan_kernel<<<dim3(NG), dim3(256), 0, stream>>>(row_cnt, row_off);

  // 5) scatter edge values by row-major ordinal
  scatter_kernel<<<dim3(NG * AN), dim3(256), 0, stream>>>(adj, row_off, t_out, pred_adj);

  // 6) embeddings = x
  const int n4 = NG * AN * DF / 4;
  copy_kernel<<<dim3((n4 + 255) / 256), dim3(256), 0, stream>>>(
      (const float4*)x, (float4*)emb, n4);
}
